// Seq2SeqModel_57741540328079
// MI455X (gfx1250) — matrix operand order repaired and checked by reading, not compiled, not run on hardware
//
#include <hip/hip_runtime.h>

#define HID 1024
#define SEQ 256
#define VOC 32000
#define G3  3072     // 3*HID
#define NB  32       // persistent-kernel workgroups (co-resident, spin barrier)
#define BOS 1

typedef __attribute__((ext_vector_type(16))) __bf16 v16bf;
typedef __attribute__((ext_vector_type(8)))  float  v8f;
typedef __attribute__((ext_vector_type(4)))  unsigned int u32x4;

union ABfrag { v16bf v; u32x4 q[2]; };

__device__ __forceinline__ unsigned short f2bf(float f) {
  unsigned u = __float_as_uint(f);
  u += 0x7FFFu + ((u >> 16) & 1u);          // round-to-nearest-even
  return (unsigned short)(u >> 16);
}
__device__ __forceinline__ float bf2f(unsigned short u) {
  return __uint_as_float((unsigned)u << 16);
}
__device__ __forceinline__ float sigmoidf_(float x) { return 1.0f / (1.0f + expf(-x)); }

// ---------------------------------------------------------------------------
// Grid-wide barrier for persistent kernels (counter+generation in d_ws).
// ---------------------------------------------------------------------------
__device__ __forceinline__ void grid_sync(unsigned* cnt, unsigned* gen, unsigned nb) {
  __syncthreads();
  __threadfence();
  if (threadIdx.x == 0) {
    volatile unsigned* vgen = (volatile unsigned*)gen;
    unsigned g = *vgen;
    unsigned prev = atomicAdd(cnt, 1u);
    if (prev + 1u == nb) {
      atomicExch(cnt, 0u);
      __threadfence();
      atomicAdd(gen, 1u);
    } else {
      while (*vgen == g) { __builtin_amdgcn_s_sleep(2); }
    }
  }
  __syncthreads();
}

// ---------------------------------------------------------------------------
// fp32 -> bf16 converters / gathers
// ---------------------------------------------------------------------------
__global__ void __launch_bounds__(256) k_convert(const float* __restrict__ src,
                                                 unsigned short* __restrict__ dst,
                                                 long long n) {
  long long i = (long long)blockIdx.x * blockDim.x + threadIdx.x;
  long long stride = (long long)gridDim.x * blockDim.x;
  for (; i < n; i += stride) dst[i] = f2bf(src[i]);
}

// src is rows x 2048; split into two rows x 1024 bf16 matrices
__global__ void __launch_bounds__(256) k_split(const float* __restrict__ src,
                                               unsigned short* __restrict__ dA,
                                               unsigned short* __restrict__ dB,
                                               long long n /* rows*1024 */) {
  long long i = (long long)blockIdx.x * blockDim.x + threadIdx.x;
  long long stride = (long long)gridDim.x * blockDim.x;
  for (; i < n; i += stride) {
    long long r = i >> 10; int c = (int)(i & 1023);
    dA[i] = f2bf(src[r * 2048 + c]);
    dB[i] = f2bf(src[r * 2048 + 1024 + c]);
  }
}

// dst[t][c] = emb[tok_t][c]; shift=1 -> tok_0 = BOS, tok_t = toks[t-1]
__global__ void __launch_bounds__(256) k_gather(const float* __restrict__ emb,
                                                const int* __restrict__ toks,
                                                unsigned short* __restrict__ dst,
                                                int shift) {
  int i = blockIdx.x * blockDim.x + threadIdx.x;        // exactly SEQ*HID threads
  int t = i >> 10, c = i & 1023;
  int tok = shift ? (t == 0 ? BOS : toks[t - 1]) : toks[t];
  dst[i] = f2bf(emb[(size_t)tok * HID + c]);
}

__global__ void __launch_bounds__(256) k_init(float* __restrict__ hbuf,
                                              unsigned* __restrict__ sync) {
  int i = blockIdx.x * blockDim.x + threadIdx.x;
  if (i < 2 * HID) hbuf[i] = 0.0f;
  if (i < 8) sync[i] = 0u;
}

// ---------------------------------------------------------------------------
// WMMA bf16 GEMM:  C[m][n] = sum_k A[m][k] * B[n][k]   (both row-major, fp32 out)
// One 16x16 output tile per wave, K in chunks of 32, v_wmma_f32_16x16x32_bf16.
// ---------------------------------------------------------------------------
__global__ void __launch_bounds__(256) k_gemm_bf16(const unsigned short* __restrict__ A,
                                                   const unsigned short* __restrict__ B,
                                                   float* __restrict__ C,
                                                   int M, int N, int K,
                                                   int lda, int ldb, int ldc) {
  const int lane = threadIdx.x & 31;
  const int wave = threadIdx.x >> 5;
  const int tilesN = N >> 4;
  const int tile = blockIdx.x * (blockDim.x >> 5) + wave;
  if (tile >= (M >> 4) * tilesN) return;                // wave-uniform
  const int tm = (tile / tilesN) << 4;
  const int tn = (tile % tilesN) << 4;
  const int r  = lane & 15;                             // A row / B col within tile
  const int kg = lane >> 4;                             // K-half select
  // A 16x32 bf16 fragment: lane kg=0 holds K {0..7,16..23}, kg=1 holds K {8..15,24..31}
  const unsigned short* Ap = A + (size_t)(tm + r) * lda + kg * 8;
  // B 32x16 bf16 fragment: lane kg=0 holds K {0..7,16..23}, kg=1 holds K {8..15,24..31}
  const unsigned short* Bp = B + (size_t)(tn + r) * ldb + kg * 8;
  v8f acc = {0.f, 0.f, 0.f, 0.f, 0.f, 0.f, 0.f, 0.f};
  for (int k0 = 0; k0 < K; k0 += 32) {
    __builtin_prefetch(Ap + k0 + 128, 0, 1);            // global_prefetch_b8
    __builtin_prefetch(Bp + k0 + 128, 0, 1);
    ABfrag a, b;
    a.q[0] = *(const u32x4*)(Ap + k0);
    a.q[1] = *(const u32x4*)(Ap + k0 + 16);
    b.q[0] = *(const u32x4*)(Bp + k0);
    b.q[1] = *(const u32x4*)(Bp + k0 + 16);
    acc = __builtin_amdgcn_wmma_f32_16x16x32_bf16(false, a.v, false, b.v,
                                                  (short)0, acc, false, false);
  }
  // D layout: N = lane&15, M = vgpr + 8*(lane>>4)
  float* Cp = C + (size_t)(tm + kg * 8) * ldc + tn + r;
#pragma unroll
  for (int v = 0; v < 8; ++v) Cp[(size_t)v * ldc] = acc[v];
}

// ---------------------------------------------------------------------------
// Persistent encoder: per step t only gh = enc_w_hh @ h is sequential; the
// input-side gates GI = enc_w_ih @ X were precomputed by WMMA GEMM.
// 32 WGs x 256 thr; WG owns 32 hidden rows, 8 threads per row (K slices of 128).
// ---------------------------------------------------------------------------
__global__ void __launch_bounds__(256) k_encoder(const unsigned short* __restrict__ whh,
                                                 const float* __restrict__ GI,   // 3072 x 256
                                                 const float* __restrict__ b_ih,
                                                 const float* __restrict__ b_hh,
                                                 float* __restrict__ hbuf,       // 2 x 1024
                                                 float* __restrict__ enc_outs,   // 256 x 1024
                                                 unsigned* sync_cnt, unsigned* sync_gen) {
  __shared__ float red[32][8][3];
  const int lr = threadIdx.x >> 3;
  const int s  = threadIdx.x & 7;
  const int j  = blockIdx.x * 32 + lr;                  // hidden index 0..1023
  const unsigned* w0 = (const unsigned*)(whh + (size_t)j * HID + s * 128);
  const unsigned* w1 = (const unsigned*)(whh + (size_t)(j + HID) * HID + s * 128);
  const unsigned* w2 = (const unsigned*)(whh + (size_t)(j + 2 * HID) * HID + s * 128);
  for (int t = 0; t < SEQ; ++t) {
    const float* h  = hbuf + (t & 1) * HID;
    float*       hn = hbuf + ((t + 1) & 1) * HID;
    const float* hs = h + s * 128;
    float p0 = 0.f, p1 = 0.f, p2 = 0.f;
#pragma unroll 8
    for (int k = 0; k < 64; ++k) {
      unsigned a = w0[k], b = w1[k], c = w2[k];
      float h0 = hs[2 * k], h1 = hs[2 * k + 1];
      p0 = fmaf(bf2f((unsigned short)a), h0, p0); p0 = fmaf(bf2f((unsigned short)(a >> 16)), h1, p0);
      p1 = fmaf(bf2f((unsigned short)b), h0, p1); p1 = fmaf(bf2f((unsigned short)(b >> 16)), h1, p1);
      p2 = fmaf(bf2f((unsigned short)c), h0, p2); p2 = fmaf(bf2f((unsigned short)(c >> 16)), h1, p2);
    }
    red[lr][s][0] = p0; red[lr][s][1] = p1; red[lr][s][2] = p2;
    __syncthreads();
    if (s == 0) {
      float g0 = 0.f, g1 = 0.f, g2 = 0.f;
      for (int q = 0; q < 8; ++q) { g0 += red[lr][q][0]; g1 += red[lr][q][1]; g2 += red[lr][q][2]; }
      float ir = GI[(size_t)j * SEQ + t]             + b_ih[j];
      float iz = GI[(size_t)(j + HID) * SEQ + t]     + b_ih[j + HID];
      float in_= GI[(size_t)(j + 2 * HID) * SEQ + t] + b_ih[j + 2 * HID];
      float hr = g0 + b_hh[j];
      float hz = g1 + b_hh[j + HID];
      float hn_= g2 + b_hh[j + 2 * HID];
      float rg = sigmoidf_(ir + hr);
      float zg = sigmoidf_(iz + hz);
      float ng = tanhf(in_ + rg * hn_);
      float hv = (1.f - zg) * ng + zg * h[j];
      hn[j] = hv;
      enc_outs[(size_t)t * HID + j] = hv;
    }
    grid_sync(sync_cnt, sync_gen, NB);
  }
}

// ---------------------------------------------------------------------------
// Persistent decoder. Embedding-side GEMMs (A_e, C_e) precomputed with WMMA.
// Per step: attn-on-h matvec -> softmax+ctx -> comb-on-ctx -> GRU. h2 saved
// as bf16 rows (H2) for the big out_w WMMA GEMM afterwards.
// ---------------------------------------------------------------------------
__global__ void __launch_bounds__(256) k_decoder(const unsigned short* __restrict__ attn_h_bf, // 256x1024
                                                 const unsigned short* __restrict__ comb_c_bf, // 1024x1024
                                                 const unsigned short* __restrict__ dwih,      // 3072x1024
                                                 const unsigned short* __restrict__ dwhh,      // 3072x1024
                                                 const float* __restrict__ A_e,     // 256 x 256
                                                 const float* __restrict__ C_e,     // 1024 x 256
                                                 const float* __restrict__ attn_b,
                                                 const float* __restrict__ comb_b,
                                                 const float* __restrict__ dbih,
                                                 const float* __restrict__ dbhh,
                                                 const float* __restrict__ enc_outs,
                                                 float* __restrict__ hbuf,          // hbuf[0] = enc_h at entry
                                                 float* __restrict__ a_scores,      // 256
                                                 float* __restrict__ ctxv,          // 1024
                                                 float* __restrict__ o_vec,         // 1024
                                                 unsigned short* __restrict__ H2,   // 256 x 1024 bf16
                                                 unsigned* sync_cnt, unsigned* sync_gen) {
  __shared__ float red[32][8][6];
  __shared__ float sm[256];
  const int tid  = threadIdx.x;
  const int lr   = tid >> 3, s = tid & 7;
  const int j    = blockIdx.x * 32 + lr;                // hidden index
  const int lane = tid & 31, wv = tid >> 5;
  const int ai   = blockIdx.x * 8 + wv;                 // attention row 0..255
  const unsigned* awp  = (const unsigned*)(attn_h_bf + (size_t)ai * HID + lane * 32);
  const unsigned* cwp  = (const unsigned*)(comb_c_bf + (size_t)j * HID + s * 128);
  const unsigned* wi0  = (const unsigned*)(dwih + (size_t)j * HID + s * 128);
  const unsigned* wi1  = (const unsigned*)(dwih + (size_t)(j + HID) * HID + s * 128);
  const unsigned* wi2  = (const unsigned*)(dwih + (size_t)(j + 2 * HID) * HID + s * 128);
  const unsigned* wh0  = (const unsigned*)(dwhh + (size_t)j * HID + s * 128);
  const unsigned* wh1  = (const unsigned*)(dwhh + (size_t)(j + HID) * HID + s * 128);
  const unsigned* wh2  = (const unsigned*)(dwhh + (size_t)(j + 2 * HID) * HID + s * 128);

  for (int t = 0; t < SEQ; ++t) {
    const float* h  = hbuf + (t & 1) * HID;
    float*       hn = hbuf + ((t + 1) & 1) * HID;
    { // stage 1: a[i] = A_e[i][t] + attn_b[i] + attn_w_h[i] . h   (1 wave per row)
      const float* hh = h + lane * 32;
      float p = 0.f;
#pragma unroll 4
      for (int k = 0; k < 16; ++k) {
        unsigned a = awp[k];
        p = fmaf(bf2f((unsigned short)a),       hh[2 * k],     p);
        p = fmaf(bf2f((unsigned short)(a >> 16)), hh[2 * k + 1], p);
      }
#pragma unroll
      for (int off = 16; off > 0; off >>= 1) p += __shfl_down(p, off, 32);
      if (lane == 0) a_scores[ai] = A_e[(size_t)ai * SEQ + t] + attn_b[ai] + p;
    }
    grid_sync(sync_cnt, sync_gen, NB);
    { // stage 2: softmax (redundant per WG) + ctx[j] = sum_i aw[i] * enc_outs[i][j]
      float v = a_scores[tid];
      sm[tid] = v; __syncthreads();
      for (int off = 128; off > 0; off >>= 1) { if (tid < off) sm[tid] = fmaxf(sm[tid], sm[tid + off]); __syncthreads(); }
      float m = sm[0]; __syncthreads();
      float e = expf(v - m);
      sm[tid] = e; __syncthreads();
      for (int off = 128; off > 0; off >>= 1) { if (tid < off) sm[tid] += sm[tid + off]; __syncthreads(); }
      float S = sm[0]; __syncthreads();
      sm[tid] = e / S; __syncthreads();
      float p = 0.f;
      for (int ii = 0; ii < 32; ++ii) { int i = s * 32 + ii; p = fmaf(sm[i], enc_outs[(size_t)i * HID + j], p); }
      red[lr][s][0] = p; __syncthreads();
      if (s == 0) { float c = 0.f; for (int q = 0; q < 8; ++q) c += red[lr][q][0]; ctxv[j] = c; }
    }
    grid_sync(sync_cnt, sync_gen, NB);
    { // stage 3: o = relu(C_e[:,t] + comb_w_c @ ctx + comb_b)
      const float* cs = ctxv + s * 128;
      float p = 0.f;
#pragma unroll 8
      for (int k = 0; k < 64; ++k) {
        unsigned a = cwp[k];
        p = fmaf(bf2f((unsigned short)a),       cs[2 * k],     p);
        p = fmaf(bf2f((unsigned short)(a >> 16)), cs[2 * k + 1], p);
      }
      red[lr][s][0] = p; __syncthreads();
      if (s == 0) {
        float acc = C_e[(size_t)j * SEQ + t] + comb_b[j];
        for (int q = 0; q < 8; ++q) acc += red[lr][q][0];
        o_vec[j] = acc > 0.f ? acc : 0.f;
      }
    }
    grid_sync(sync_cnt, sync_gen, NB);
    { // stage 4: GRU: gi = dec_w_ih @ o, gh = dec_w_hh @ h, gates, h2
      const float* os = o_vec + s * 128;
      const float* hs = h + s * 128;
      float q0=0.f,q1=0.f,q2=0.f,q3=0.f,q4=0.f,q5=0.f;
#pragma unroll 4
      for (int k = 0; k < 64; ++k) {
        float o0 = os[2*k], o1 = os[2*k+1], h0 = hs[2*k], h1 = hs[2*k+1];
        unsigned a;
        a = wi0[k]; q0 = fmaf(bf2f((unsigned short)a), o0, q0); q0 = fmaf(bf2f((unsigned short)(a>>16)), o1, q0);
        a = wi1[k]; q1 = fmaf(bf2f((unsigned short)a), o0, q1); q1 = fmaf(bf2f((unsigned short)(a>>16)), o1, q1);
        a = wi2[k]; q2 = fmaf(bf2f((unsigned short)a), o0, q2); q2 = fmaf(bf2f((unsigned short)(a>>16)), o1, q2);
        a = wh0[k]; q3 = fmaf(bf2f((unsigned short)a), h0, q3); q3 = fmaf(bf2f((unsigned short)(a>>16)), h1, q3);
        a = wh1[k]; q4 = fmaf(bf2f((unsigned short)a), h0, q4); q4 = fmaf(bf2f((unsigned short)(a>>16)), h1, q4);
        a = wh2[k]; q5 = fmaf(bf2f((unsigned short)a), h0, q5); q5 = fmaf(bf2f((unsigned short)(a>>16)), h1, q5);
      }
      red[lr][s][0]=q0; red[lr][s][1]=q1; red[lr][s][2]=q2;
      red[lr][s][3]=q3; red[lr][s][4]=q4; red[lr][s][5]=q5;
      __syncthreads();
      if (s == 0) {
        float g[6] = {0.f,0.f,0.f,0.f,0.f,0.f};
        for (int q = 0; q < 8; ++q)
#pragma unroll
          for (int x = 0; x < 6; ++x) g[x] += red[lr][q][x];
        float ir = g[0] + dbih[j], iz = g[1] + dbih[j + HID], inn = g[2] + dbih[j + 2 * HID];
        float hr = g[3] + dbhh[j], hz = g[4] + dbhh[j + HID], hnn = g[5] + dbhh[j + 2 * HID];
        float rg = sigmoidf_(ir + hr);
        float zg = sigmoidf_(iz + hz);
        float ng = tanhf(inn + rg * hnn);
        float h2 = (1.f - zg) * ng + zg * h[j];
        hn[j] = h2;
        H2[(size_t)t * HID + j] = f2bf(h2);
      }
    }
    grid_sync(sync_cnt, sync_gen, NB);
  }
}

// ---------------------------------------------------------------------------
// loss_t = -(v[tgt] - max - log sum exp(v - max)),  v = logitsT[t] + out_b
// ---------------------------------------------------------------------------
__global__ void __launch_bounds__(256) k_loss(const float* __restrict__ logitsT,
                                              const float* __restrict__ out_b,
                                              const int* __restrict__ target,
                                              float* __restrict__ lossbuf) {
  __shared__ float sm[256];
  const int t = blockIdx.x;
  const float* row = logitsT + (size_t)t * VOC;
  float m = -3.4e38f;
  for (int jj = threadIdx.x; jj < VOC; jj += 256) m = fmaxf(m, row[jj] + out_b[jj]);
  sm[threadIdx.x] = m; __syncthreads();
  for (int off = 128; off > 0; off >>= 1) { if (threadIdx.x < off) sm[threadIdx.x] = fmaxf(sm[threadIdx.x], sm[threadIdx.x + off]); __syncthreads(); }
  m = sm[0]; __syncthreads();
  float ssum = 0.f;
  for (int jj = threadIdx.x; jj < VOC; jj += 256) ssum += expf(row[jj] + out_b[jj] - m);
  sm[threadIdx.x] = ssum; __syncthreads();
  for (int off = 128; off > 0; off >>= 1) { if (threadIdx.x < off) sm[threadIdx.x] += sm[threadIdx.x + off]; __syncthreads(); }
  if (threadIdx.x == 0) {
    int tg = target[t];
    float v = row[tg] + out_b[tg];
    lossbuf[t] = -(v - m - logf(sm[0]));
  }
}

__global__ void __launch_bounds__(256) k_sum(const float* __restrict__ lossbuf,
                                             float* __restrict__ out) {
  __shared__ float sm[256];
  sm[threadIdx.x] = lossbuf[threadIdx.x]; __syncthreads();
  for (int off = 128; off > 0; off >>= 1) { if (threadIdx.x < off) sm[threadIdx.x] += sm[threadIdx.x + off]; __syncthreads(); }
  if (threadIdx.x == 0) out[0] = sm[0];
}

// ---------------------------------------------------------------------------
extern "C" void kernel_launch(void* const* d_in, const int* in_sizes, int n_in,
                              void* d_out, int out_size, void* d_ws, size_t ws_size,
                              hipStream_t stream) {
  (void)in_sizes; (void)n_in; (void)out_size; (void)ws_size;
  const int*   input_t  = (const int*)d_in[0];
  const int*   target_t = (const int*)d_in[1];
  const float* enc_emb  = (const float*)d_in[2];
  const float* enc_w_ih = (const float*)d_in[3];
  const float* enc_w_hh = (const float*)d_in[4];
  const float* enc_b_ih = (const float*)d_in[5];
  const float* enc_b_hh = (const float*)d_in[6];
  const float* dec_emb  = (const float*)d_in[7];
  const float* attn_w   = (const float*)d_in[8];
  const float* attn_b   = (const float*)d_in[9];
  const float* comb_w   = (const float*)d_in[10];
  const float* comb_b   = (const float*)d_in[11];
  const float* dec_w_ih = (const float*)d_in[12];
  const float* dec_w_hh = (const float*)d_in[13];
  const float* dec_b_ih = (const float*)d_in[14];
  const float* dec_b_hh = (const float*)d_in[15];
  const float* out_w    = (const float*)d_in[16];
  const float* out_b    = (const float*)d_in[17];

  char* base = (char*)d_ws;
  size_t off = 0;
  auto take = [&](size_t bytes) -> void* {
    void* p = base + off;
    off = (off + bytes + 255) & ~(size_t)255;
    return p;
  };
  unsigned short* enc_wih_bf = (unsigned short*)take((size_t)G3 * HID * 2);
  unsigned short* enc_whh_bf = (unsigned short*)take((size_t)G3 * HID * 2);
  unsigned short* dec_wih_bf = (unsigned short*)take((size_t)G3 * HID * 2);
  unsigned short* dec_whh_bf = (unsigned short*)take((size_t)G3 * HID * 2);
  unsigned short* comb_e_bf  = (unsigned short*)take((size_t)HID * HID * 2);
  unsigned short* comb_c_bf  = (unsigned short*)take((size_t)HID * HID * 2);
  unsigned short* attn_e_bf  = (unsigned short*)take((size_t)SEQ * HID * 2);
  unsigned short* attn_h_bf  = (unsigned short*)take((size_t)SEQ * HID * 2);
  unsigned short* outw_bf    = (unsigned short*)take((size_t)VOC * HID * 2);
  unsigned short* X_bf       = (unsigned short*)take((size_t)SEQ * HID * 2);
  unsigned short* E_bf       = (unsigned short*)take((size_t)SEQ * HID * 2);
  unsigned short* H2_bf      = (unsigned short*)take((size_t)SEQ * HID * 2);
  float* GI       = (float*)take((size_t)G3 * SEQ * 4);
  float* A_e      = (float*)take((size_t)SEQ * SEQ * 4);
  float* C_e      = (float*)take((size_t)HID * SEQ * 4);
  float* enc_outs = (float*)take((size_t)SEQ * HID * 4);
  float* hbuf     = (float*)take((size_t)2 * HID * 4);
  float* a_scores = (float*)take((size_t)SEQ * 4);
  float* ctxv     = (float*)take((size_t)HID * 4);
  float* o_vec    = (float*)take((size_t)HID * 4);
  float* logitsT  = (float*)take((size_t)SEQ * VOC * 4);
  float* lossbuf  = (float*)take((size_t)SEQ * 4);
  unsigned* sync  = (unsigned*)take(64);

  // 0) init state (h0 = 0, barrier counters = 0) — every call (graph replays)
  k_init<<<9, 256, 0, stream>>>(hbuf, sync);

  // 1) one-time (per launch) fp32 -> bf16 conversions + embedding gathers
  k_convert<<<2048, 256, 0, stream>>>(enc_w_ih, enc_wih_bf, (long long)G3 * HID);
  k_convert<<<2048, 256, 0, stream>>>(enc_w_hh, enc_whh_bf, (long long)G3 * HID);
  k_convert<<<2048, 256, 0, stream>>>(dec_w_ih, dec_wih_bf, (long long)G3 * HID);
  k_convert<<<2048, 256, 0, stream>>>(dec_w_hh, dec_whh_bf, (long long)G3 * HID);
  k_convert<<<8192, 256, 0, stream>>>(out_w, outw_bf, (long long)VOC * HID);
  k_split<<<2048, 256, 0, stream>>>(comb_w, comb_e_bf, comb_c_bf, (long long)HID * HID);
  k_split<<<512, 256, 0, stream>>>(attn_w, attn_e_bf, attn_h_bf, (long long)SEQ * HID);
  k_gather<<<(SEQ * HID) / 256, 256, 0, stream>>>(enc_emb, input_t, X_bf, 0);
  k_gather<<<(SEQ * HID) / 256, 256, 0, stream>>>(dec_emb, target_t, E_bf, 1);

  // 2) batched input-side GEMMs (WMMA bf16): embeddings are known ahead of time
  //    GI  = enc_w_ih @ X^T   (3072 x 256)
  k_gemm_bf16<<<(G3 / 16) * (SEQ / 16) / 8, 256, 0, stream>>>(
      enc_wih_bf, X_bf, GI, G3, SEQ, HID, HID, HID, SEQ);
  //    A_e = attn_w[:, :H] @ E^T   (256 x 256)
  k_gemm_bf16<<<(SEQ / 16) * (SEQ / 16) / 8, 256, 0, stream>>>(
      attn_e_bf, E_bf, A_e, SEQ, SEQ, HID, HID, HID, SEQ);
  //    C_e = comb_w[:, :H] @ E^T   (1024 x 256)
  k_gemm_bf16<<<(HID / 16) * (SEQ / 16) / 8, 256, 0, stream>>>(
      comb_e_bf, E_bf, C_e, HID, SEQ, HID, HID, HID, SEQ);

  // 3) sequential phases: persistent kernels with grid barrier
  k_encoder<<<NB, 256, 0, stream>>>(enc_whh_bf, GI, enc_b_ih, enc_b_hh,
                                    hbuf, enc_outs, sync + 0, sync + 1);
  k_decoder<<<NB, 256, 0, stream>>>(attn_h_bf, comb_c_bf, dec_wih_bf, dec_whh_bf,
                                    A_e, C_e, attn_b, comb_b, dec_b_ih, dec_b_hh,
                                    enc_outs, hbuf, a_scores, ctxv, o_vec, H2_bf,
                                    sync + 2, sync + 3);

  // 4) hoisted output projection as one big WMMA GEMM:
  //    logitsT(256 x 32000) = H2(256 x 1024) @ out_w^T
  k_gemm_bf16<<<(SEQ / 16) * (VOC / 16) / 8, 256, 0, stream>>>(
      H2_bf, outw_bf, logitsT, SEQ, VOC, HID, HID, HID, VOC);

  // 5) per-step logsumexp loss + final reduction
  k_loss<<<SEQ, 256, 0, stream>>>(logitsT, out_b, target_t, lossbuf);
  k_sum<<<1, 256, 0, stream>>>(lossbuf, (float*)d_out);
}
